// GroupRouter_5935644803719
// MI455X (gfx1250) — matrix operand order repaired and checked by reading, not compiled, not run on hardware
//
#include <hip/hip_runtime.h>

// ---- CDNA5 WMMA / vector types ----
typedef __attribute__((ext_vector_type(16))) _Float16 v16h;
typedef __attribute__((ext_vector_type(8)))  float    v8f;
typedef __attribute__((ext_vector_type(4)))  float    v4f;   // clang vector: legal for nontemporal builtins

// Problem constants (match the reference)
#define DIM   512
#define NG    4          // groups
#define NGS   4          // group size
#define NOUT  20         // 4 group logits + 16 in-group logits
#define NPAD  32         // padded to two 16-wide WMMA N tiles
#define TOPK  2
#define KCHUNKS (DIM / 32)   // 16 chunks of K=32

// One 16-token tile per wave, 8 waves (256 threads) per block -> 128 tokens/block.
__global__ __launch_bounds__(256)
void group_router_wmma_kernel(const float* __restrict__ x,        // [B,512]
                              const float* __restrict__ group_w,  // [4,512]
                              const float* __restrict__ group_b,  // [4]
                              const float* __restrict__ in_w,     // [4,4,512]
                              const float* __restrict__ in_b,     // [4,4]
                              const int*   __restrict__ experts,  // [16]
                              float* __restrict__ out,            // [2*B] idx-as-float, then [2*B] weights
                              long Btot) {
    // B-matrix fragments, pre-swizzled: [kchunk][ntile][lane] of v16h (32B each) = 32 KB
    __shared__ v16h  sB[KCHUNKS][2][32];
    __shared__ float sBias[NPAD];
    __shared__ int   sExp[NG * NGS];
    __shared__ float sLog[8][16][NOUT];   // per-wave logit tile [token][n]

    const int tid = threadIdx.x;

    // ---- Stage 0: build f16 B fragments in LDS ------------------------------
    // Dense 16-bit B (32x16, KxN) lane layout used here (contiguous K, same order as A):
    //   lanes 0-15 hold K=0..15 (VGPR v -> K=2v,2v+1), lanes 16-31 hold K=16..31.
    // Flat element idx = ((c*2 + t)*32 + l)*16 + e ; e indexes the 16 halves.
    _Float16* sBh = (_Float16*)sB;
    for (int idx = tid; idx < KCHUNKS * 2 * 32 * 16; idx += 256) {
        const int e = idx & 15;
        const int l = (idx >> 4) & 31;
        const int t = (idx >> 9) & 1;
        const int c = idx >> 10;
        const int k = c * 32 + ((l < 16) ? e : (16 + e));
        const int n = (l & 15) + 16 * t;
        float v = 0.f;
        if (n < NG)            v = group_w[n * DIM + k];
        else if (n < NOUT)     v = in_w[(n - NG) * DIM + k];
        sBh[idx] = (_Float16)v;
    }
    if (tid < NPAD) {
        float bv = 0.f;
        if (tid < NG)         bv = group_b[tid];
        else if (tid < NOUT)  bv = in_b[tid - NG];
        sBias[tid] = bv;
    }
    if (tid < NG * NGS) sExp[tid] = experts[tid];
    __syncthreads();

    // ---- Main: 16-token x 32-output GEMM on WMMA ----------------------------
    const int wave = tid >> 5;
    const int lane = tid & 31;
    const long tokenBase = (long)blockIdx.x * 256 + wave * 16;
    const int  m    = lane & 15;                 // A-matrix row for this lane
    const int  off1 = (lane < 16) ? 0 : 16;      // K sub-offsets: contiguous K=16h..16h+15
    const int  off2 = off1 + 8;

    const float* xrow = x + (tokenBase + m) * (long)DIM;

    v8f c0 = {}, c1 = {};
    #pragma unroll
    for (int c = 0; c < KCHUNKS; ++c) {
        // 16 f32 from this lane's token row -> packed f16 A fragment.
        // x is a 512 MiB single-pass stream: non-temporal (TH=NT) keeps it from
        // evicting the hot 40 KB weight table (reused by every block) out of L2.
        const v4f* p1 = (const v4f*)(xrow + c * 32 + off1);
        const v4f* p2 = (const v4f*)(xrow + c * 32 + off2);
        const v4f a0 = __builtin_nontemporal_load(p1);
        const v4f a1 = __builtin_nontemporal_load(p1 + 1);
        const v4f b0 = __builtin_nontemporal_load(p2);
        const v4f b1 = __builtin_nontemporal_load(p2 + 1);
        v16h A;
        A[0]  = (_Float16)a0.x; A[1]  = (_Float16)a0.y;
        A[2]  = (_Float16)a0.z; A[3]  = (_Float16)a0.w;
        A[4]  = (_Float16)a1.x; A[5]  = (_Float16)a1.y;
        A[6]  = (_Float16)a1.z; A[7]  = (_Float16)a1.w;
        A[8]  = (_Float16)b0.x; A[9]  = (_Float16)b0.y;
        A[10] = (_Float16)b0.z; A[11] = (_Float16)b0.w;
        A[12] = (_Float16)b1.x; A[13] = (_Float16)b1.y;
        A[14] = (_Float16)b1.z; A[15] = (_Float16)b1.w;

        const v16h B0 = sB[c][0][lane];
        const v16h B1 = sB[c][1][lane];
        // D = A x B + C, f32 accumulate
        c0 = __builtin_amdgcn_wmma_f32_16x16x32_f16(false, A, false, B0,
                                                    (short)0, c0, false, false);
        c1 = __builtin_amdgcn_wmma_f32_16x16x32_f16(false, A, false, B1,
                                                    (short)0, c1, false, false);
    }

    // ---- Scatter C (+bias) to per-wave logit tile ---------------------------
    // C layout: lane l holds column N=(l&15); VGPR r holds row M=r (lanes<16) / r+8.
    const int nbase = lane & 15;
    #pragma unroll
    for (int r = 0; r < 8; ++r) {
        const int M = r + ((lane < 16) ? 0 : 8);
        sLog[wave][M][nbase] = c0[r] + sBias[nbase];
        if (nbase < NOUT - 16)
            sLog[wave][M][16 + nbase] = c1[r] + sBias[16 + nbase];
    }
    __syncthreads();

    // ---- Routing epilogue: one lane per token -------------------------------
    if (lane < 16) {
        const long  token = tokenBase + lane;
        const float* L = sLog[wave][lane];

        // argmax over group logits (first-max, matches jnp.argmax)
        int g = 0; float best = L[0];
        #pragma unroll
        for (int j = 1; j < NG; ++j) if (L[j] > best) { best = L[j]; g = j; }

        // softmax over the chosen group's in-group logits
        float gl[NGS];
        #pragma unroll
        for (int j = 0; j < NGS; ++j) gl[j] = L[NG + NGS * g + j];
        float mx = gl[0];
        #pragma unroll
        for (int j = 1; j < NGS; ++j) mx = fmaxf(mx, gl[j]);
        float p[NGS]; float s = 0.f;
        #pragma unroll
        for (int j = 0; j < NGS; ++j) { p[j] = __expf(gl[j] - mx); s += p[j]; }
        const float inv = 1.f / s;
        #pragma unroll
        for (int j = 0; j < NGS; ++j) p[j] *= inv;

        // top-2 (descending, first-index tie-break like lax.top_k)
        int i0 = 0; float w0 = p[0];
        #pragma unroll
        for (int j = 1; j < NGS; ++j) if (p[j] > w0) { w0 = p[j]; i0 = j; }
        int i1 = -1; float w1 = -1.f;
        #pragma unroll
        for (int j = 0; j < NGS; ++j)
            if (j != i0 && p[j] > w1) { w1 = p[j]; i1 = j; }

        const int e0 = sExp[g * NGS + i0];
        const int e1 = sExp[g * NGS + i1];

        // outputs: expert indices (as float) then weights, written once and
        // never re-read by this kernel -> non-temporal stores.
        __builtin_nontemporal_store((float)e0, out + token * TOPK + 0);
        __builtin_nontemporal_store((float)e1, out + token * TOPK + 1);
        __builtin_nontemporal_store(w0, out + Btot * TOPK + token * TOPK + 0);
        __builtin_nontemporal_store(w1, out + Btot * TOPK + token * TOPK + 1);
    }
}

extern "C" void kernel_launch(void* const* d_in, const int* in_sizes, int n_in,
                              void* d_out, int out_size, void* d_ws, size_t ws_size,
                              hipStream_t stream) {
    const float* x        = (const float*)d_in[0];  // routing_features [B,512]
    const float* group_w  = (const float*)d_in[1];  // [4,512]
    const float* group_b  = (const float*)d_in[2];  // [4]
    const float* in_w     = (const float*)d_in[3];  // [4,4,512]
    const float* in_b     = (const float*)d_in[4];  // [4,4]
    const int*   experts  = (const int*)d_in[5];    // [16]
    float*       out      = (float*)d_out;

    const long Btot = (long)in_sizes[0] / DIM;      // 262144
    const int  blocks = (int)(Btot / 256);          // 256-token stride per block

    group_router_wmma_kernel<<<blocks, 256, 0, stream>>>(
        x, group_w, group_b, in_w, in_b, experts, out, Btot);
}
